// MultiHeadedAttention_29119878267636
// MI455X (gfx1250) — compile-verified, matrix operand order checked by reading, not run on hardware
//
#include <hip/hip_runtime.h>
#include <hip/hip_bf16.h>

// ---------------- problem constants (from reference) ----------------
#define BATCH   2
#define SEQ     2048
#define DMODEL  1024
#define NHEAD   16
#define HDIM    64
#define KDIM    1024          // contraction dim of the projection GEMMs (== DMODEL)
#define MROWS   (BATCH*SEQ)   // 4096

typedef __bf16 bf16_t;
typedef __attribute__((ext_vector_type(16))) __bf16 v16bf;
typedef __attribute__((ext_vector_type(8)))  __bf16 v8bf;
typedef __attribute__((ext_vector_type(4)))  __bf16 v4bf;
typedef __attribute__((ext_vector_type(8)))  float  v8f;
typedef __attribute__((ext_vector_type(4)))  float  v4f;
typedef __attribute__((ext_vector_type(4)))  unsigned int v4u;
typedef __attribute__((ext_vector_type(8)))  int    v8i;
typedef __attribute__((ext_vector_type(4)))  int    v4i;

#if defined(__has_builtin)
#if __has_builtin(__builtin_amdgcn_tensor_load_to_lds)
#define HAVE_TDM 1
#endif
#endif

// ---------------------------------------------------------------
// WMMA helper: D = A(16x32 bf16) * B(32x16 bf16) + C(16x16 f32)
// ---------------------------------------------------------------
__device__ __forceinline__ v8f wmma_bf16(v16bf a, v16bf b, v8f c) {
  return __builtin_amdgcn_wmma_f32_16x16x32_bf16(
      /*neg_a=*/false, a, /*neg_b=*/false, b,
      /*c_mod=*/(short)0, c, /*reuse_a=*/false, /*reuse_b=*/false);
}

// ---------------------------------------------------------------
// Load one 16-bit A/B WMMA fragment (ISA 16-bit A 16x32 layout):
// lane&15 = row, K chunk start = (lane>=16 ? 8 : 0); halves [c,c+8),[c+16,c+24).
// ---------------------------------------------------------------
__device__ __forceinline__ v16bf load_frag_ab(const bf16_t* __restrict__ base,
                                              int row0, int rstride, int k0, int lane) {
  const bf16_t* p = base + (long)(row0 + (lane & 15)) * rstride
                         + (k0 + ((lane >> 4) << 3));
  union { v16bf v; v8bf h[2]; } u;
  u.h[0] = *(const v8bf*)(p);
  u.h[1] = *(const v8bf*)(p + 16);
  return u.v;
}

// ---------------------------------------------------------------
// TDM: async 2-D tile (rows x 32 bf16) load from global into LDS.
// D# built per CDNA5 ISA.
// LDS padding: pad_interval=16 DWORDs (code 3), pad_amount=4 DWORDs (code 3)
// -> LDS row pitch = 20 DWORDs = 40 halves (16B aligned, bank-conflict free).
// ---------------------------------------------------------------
#define TDM_ROW_PITCH 40   // halves per LDS row after padding

#if HAVE_TDM
__device__ __forceinline__ void tdm_load_tile_2d(unsigned int lds_off,
                                                 const bf16_t* gptr,
                                                 unsigned int rows,
                                                 unsigned int row_stride_elems) {
  unsigned long long ga = (unsigned long long)(size_t)gptr;
  v4u g0;
  g0[0] = 1u;                                      // count=1, user descriptor
  g0[1] = lds_off;                                 // lds_addr (bytes)
  g0[2] = (unsigned int)(ga & 0xFFFFFFFFu);        // global_addr[31:0]
  g0[3] = (unsigned int)((ga >> 32) & 0x1FFFFFFu)  // global_addr[56:32]
        | (2u << 30);                              // type = 2 ("image")
  v8i g1;
  // data_size=1 (2B) | pad_enable | pad_interval=3 (16 DW) | pad_amount=3 (4 DW)
  g1[0] = (int)((1u << 16) | (1u << 20) | (3u << 22) | (3u << 25));
  unsigned int td0 = 1u << 20, td1 = 1u << 20;     // huge tensor dims: no OOB clip
  g1[1] = (int)((td0 & 0xFFFFu) << 16);            // tensor_dim0[15:0]
  g1[2] = (int)((td0 >> 16) | ((td1 & 0xFFFFu) << 16));
  g1[3] = (int)((td1 >> 16) | (32u << 16));        // tile_dim0 = 32 elements
  g1[4] = (int)(rows);                             // tile_dim1 = rows, tile_dim2 = 0
  g1[5] = (int)row_stride_elems;                   // tensor_dim0_stride[31:0]
  g1[6] = 0;                                       // stride0[47:32], stride1[15:0]
  g1[7] = 0;
  v4i z4 = {0, 0, 0, 0};
  v8i z8 = {0, 0, 0, 0, 0, 0, 0, 0};
  // 6-arg toolchain form: (g0, g1, g2, g3, extra, cpol)
  __builtin_amdgcn_tensor_load_to_lds(g0, g1, z4, z4, z8, 0);
}
#endif

// cooperative fallback / portability path: copy rows x 32 bf16 tile into padded LDS
__device__ __forceinline__ void coop_load_tile_2d(bf16_t* lds, const bf16_t* gptr,
                                                  int rows, int row_stride_elems,
                                                  int tid, int nthreads) {
  int chunks = rows * 4;                           // 4 x v8bf per row
  for (int c = tid; c < chunks; c += nthreads) {
    int r = c >> 2, q = c & 3;
    *(v8bf*)(lds + r * TDM_ROW_PITCH + q * 8) =
        *(const v8bf*)(gptr + (long)r * row_stride_elems + q * 8);
  }
}

// ---------------------------------------------------------------
// fp32 -> bf16 cast, 4 elements / thread
// ---------------------------------------------------------------
__global__ __launch_bounds__(256) void cast_f32_bf16_x4(const float* __restrict__ in,
                                                        bf16_t* __restrict__ out, int n4) {
  int i = blockIdx.x * blockDim.x + threadIdx.x;
  if (i < n4) {
    v4f f = ((const v4f*)in)[i];
    v4bf o;
    #pragma unroll
    for (int j = 0; j < 4; ++j) o[j] = (bf16_t)f[j];
    ((v4bf*)out)[i] = o;
  }
}

// ---------------------------------------------------------------
// Vt[b,h,d,s] = V[b,s, h*HDIM+d]
// ---------------------------------------------------------------
__global__ __launch_bounds__(256) void transpose_v_kernel(const bf16_t* __restrict__ Vb,
                                                          bf16_t* __restrict__ Vt) {
  long idx = (long)blockIdx.x * blockDim.x + threadIdx.x; // B*NHEAD*HDIM*SEQ
  int s = (int)(idx % SEQ);
  long r = idx / SEQ;
  int d = (int)(r % HDIM); r /= HDIM;
  int h = (int)(r % NHEAD);
  int b = (int)(r / NHEAD);
  Vt[idx] = Vb[((long)b * SEQ + s) * DMODEL + h * HDIM + d];
}

// ---------------------------------------------------------------
// GEMM-NT: Y[m,n] = sum_k X[m,k] * W[n,k]   (y = x @ W.T)
// Block = 128 threads (4 waves), tile 64(M) x 64(N).
// A/B 64x32 K-tiles staged in LDS by the Tensor Data Mover,
// double-buffered: DMA of tile i+1 overlaps WMMA on tile i.
// ---------------------------------------------------------------
template <bool OUT_F32>
__global__ __launch_bounds__(128) void gemm_nt_kernel(const bf16_t* __restrict__ X,
                                                      const bf16_t* __restrict__ W,
                                                      bf16_t* __restrict__ Ybf,
                                                      float*  __restrict__ Yf) {
  const int lane = threadIdx.x & 31;
  const int wave = threadIdx.x >> 5;                 // 0..3
  const int n0 = blockIdx.x * 64;
  const int m0 = blockIdx.y * 64;                    // block M base
  const int mw = wave * 16;                          // wave's row offset in tile

  __shared__ __align__(16) bf16_t sA[2][64 * TDM_ROW_PITCH];
  __shared__ __align__(16) bf16_t sB[2][64 * TDM_ROW_PITCH];

  const bf16_t* Abase = X + (long)m0 * KDIM;
  const bf16_t* Bbase = W + (long)n0 * KDIM;

#if HAVE_TDM
  unsigned int ldsA[2] = { (unsigned int)(size_t)(void*)&sA[0][0],
                           (unsigned int)(size_t)(void*)&sA[1][0] };
  unsigned int ldsB[2] = { (unsigned int)(size_t)(void*)&sB[0][0],
                           (unsigned int)(size_t)(void*)&sB[1][0] };
  if (wave == 0) {                                   // TDM ignores EXEC; one issue/wave
    tdm_load_tile_2d(ldsA[0], Abase, 64, KDIM);
    tdm_load_tile_2d(ldsB[0], Bbase, 64, KDIM);
  }
#else
  coop_load_tile_2d(&sA[0][0], Abase, 64, KDIM, threadIdx.x, 128);
  coop_load_tile_2d(&sB[0][0], Bbase, 64, KDIM, threadIdx.x, 128);
#endif

  v8f acc[4] = {};

  for (int it = 0; it < KDIM / 32; ++it) {
#if HAVE_TDM
    if (wave == 0) __builtin_amdgcn_s_wait_tensorcnt(0);  // buffers for iter `it` done
#endif
    __syncthreads();   // all waves: tile `it` visible; all done reading tile `it-1`
    if (it + 1 < KDIM / 32) {
      const int kn = (it + 1) * 32;
#if HAVE_TDM
      if (wave == 0) {
        tdm_load_tile_2d(ldsA[(it + 1) & 1], Abase + kn, 64, KDIM);
        tdm_load_tile_2d(ldsB[(it + 1) & 1], Bbase + kn, 64, KDIM);
      }
#endif
    }

    const bf16_t* Abuf = &sA[it & 1][0];
    const bf16_t* Bbuf = &sB[it & 1][0];
    // batch all fragment loads, then run the WMMAs (amortize one ds wait)
    v16bf a = load_frag_ab(Abuf, mw, TDM_ROW_PITCH, 0, lane);
    v16bf bfr[4];
    #pragma unroll
    for (int ai = 0; ai < 4; ++ai)
      bfr[ai] = load_frag_ab(Bbuf, 16 * ai, TDM_ROW_PITCH, 0, lane);
    #pragma unroll
    for (int ai = 0; ai < 4; ++ai)
      acc[ai] = wmma_bf16(a, bfr[ai], acc[ai]);

#if !HAVE_TDM
    __syncthreads();
    if (it + 1 < KDIM / 32) {
      const int kn = (it + 1) * 32;
      coop_load_tile_2d(&sA[(it + 1) & 1][0], Abase + kn, 64, KDIM, threadIdx.x, 128);
      coop_load_tile_2d(&sB[(it + 1) & 1][0], Bbase + kn, 64, KDIM, threadIdx.x, 128);
    }
#endif
  }

  // C/D layout: reg v, lane: M = v + (lane>=16 ? 8:0), N = lane&15
  const int mrow = (lane >> 4) << 3;
  const int ncol = lane & 15;
  #pragma unroll
  for (int ai = 0; ai < 4; ++ai)
    #pragma unroll
    for (int v = 0; v < 8; ++v) {
      long m = m0 + mw + v + mrow;
      long n = n0 + 16 * ai + ncol;
      if (OUT_F32) Yf[m * DMODEL + n] = acc[ai][v];
      else         Ybf[m * DMODEL + n] = (bf16_t)acc[ai][v];
    }
}

// ---------------------------------------------------------------
// Flash attention for one (b, h, 64-row Q tile).
// Block = 128 threads (4 waves); each wave owns 16 query rows.
// ---------------------------------------------------------------
__global__ __launch_bounds__(128) void flash_attn_kernel(const bf16_t* __restrict__ Qb,
                                                         const bf16_t* __restrict__ Kb,
                                                         const bf16_t* __restrict__ Vt,
                                                         const int*    __restrict__ mask,
                                                         bf16_t* __restrict__ Ob) {
  const int lane = threadIdx.x & 31;
  const int wave = threadIdx.x >> 5;
  const int b = blockIdx.z, h = blockIdx.y;
  const int m0 = blockIdx.x * 64 + wave * 16;   // query row tile within SEQ

  const bf16_t* Qh = Qb + (long)b * SEQ * DMODEL + h * HDIM;   // row stride DMODEL
  const bf16_t* Kh = Kb + (long)b * SEQ * DMODEL + h * HDIM;   // row stride DMODEL
  const bf16_t* Vh = Vt + (long)(b * NHEAD + h) * HDIM * SEQ;  // row (=d) stride SEQ
  const int*    Mh = mask + (long)b * SEQ * SEQ;

  __shared__ __align__(16) bf16_t lds_p[4][16 * 64];  // per-wave P staging (8 KB)
  bf16_t* myP = &lds_p[wave][0];

  // Q fragments are loop-invariant: 2 K-chunks over HDIM=64
  v16bf qa0 = load_frag_ab(Qh, m0, DMODEL, 0, lane);
  v16bf qa1 = load_frag_ab(Qh, m0, DMODEL, 32, lane);

  v8f o_acc[4] = {};
  float m_run[8], l_run[8];
  #pragma unroll
  for (int v = 0; v < 8; ++v) { m_run[v] = -1e30f; l_run[v] = 0.0f; }

  const int mrow = (lane >> 4) << 3;
  const int ncol = lane & 15;
  const float scale = 0.125f;  // 1/sqrt(64)

  for (int t0 = 0; t0 < SEQ; t0 += 64) {
    // prefetch next K tile rows (global_prefetch_b8)
    if (t0 + 64 < SEQ)
      __builtin_prefetch(Kh + (long)(t0 + 64 + (lane & 15)) * DMODEL, 0, 1);

    // ---- scores = Q @ K^T : B[d,t] = K[t,d], K rows are d-contiguous ----
    // batch fragment loads per K-chunk, then the 4 WMMAs
    v8f s_acc[4];
    {
      v16bf kb[4];
      #pragma unroll
      for (int ai = 0; ai < 4; ++ai)
        kb[ai] = load_frag_ab(Kh, t0 + 16 * ai, DMODEL, 0, lane);
      #pragma unroll
      for (int ai = 0; ai < 4; ++ai) {
        v8f z = {};
        s_acc[ai] = wmma_bf16(qa0, kb[ai], z);
      }
      #pragma unroll
      for (int ai = 0; ai < 4; ++ai)
        kb[ai] = load_frag_ab(Kh, t0 + 16 * ai, DMODEL, 32, lane);
      #pragma unroll
      for (int ai = 0; ai < 4; ++ai)
        s_acc[ai] = wmma_bf16(qa1, kb[ai], s_acc[ai]);
    }

    // ---- scale + mask, in place (keeps VGPR pressure low) ----
    #pragma unroll
    for (int ai = 0; ai < 4; ++ai)
      #pragma unroll
      for (int v = 0; v < 8; ++v) {
        int mg = m0 + v + mrow;
        int tg = t0 + 16 * ai + ncol;
        int mk = Mh[(long)mg * SEQ + tg];
        s_acc[ai][v] = (mk == 0) ? -1e9f : s_acc[ai][v] * scale;
      }

    // ---- online softmax per row (rows live in 16-lane groups) ----
    #pragma unroll
    for (int v = 0; v < 8; ++v) {
      float mt = fmaxf(fmaxf(s_acc[0][v], s_acc[1][v]),
                       fmaxf(s_acc[2][v], s_acc[3][v]));
      #pragma unroll
      for (int off = 1; off < 16; off <<= 1)
        mt = fmaxf(mt, __shfl_xor(mt, off, 32));
      float mn = fmaxf(m_run[v], mt);
      float alpha = __expf(m_run[v] - mn);
      m_run[v] = mn;
      float rsum = 0.0f;
      #pragma unroll
      for (int ai = 0; ai < 4; ++ai) {
        float p = __expf(s_acc[ai][v] - mn);
        s_acc[ai][v] = p;
        rsum += p;
      }
      #pragma unroll
      for (int off = 1; off < 16; off <<= 1)
        rsum += __shfl_xor(rsum, off, 32);
      l_run[v] = l_run[v] * alpha + rsum;
      #pragma unroll
      for (int ai = 0; ai < 4; ++ai) o_acc[ai][v] *= alpha;
    }

    // ---- reshape P: C-layout -> row-major 16x64 in LDS -> A-fragments ----
    #pragma unroll
    for (int ai = 0; ai < 4; ++ai)
      #pragma unroll
      for (int v = 0; v < 8; ++v)
        myP[(v + mrow) * 64 + 16 * ai + ncol] = (bf16_t)s_acc[ai][v];
    // same-wave DS ops are ordered; compiler inserts s_wait_dscnt
    v16bf pa0 = load_frag_ab(myP, 0, 64, 0, lane);
    v16bf pa1 = load_frag_ab(myP, 0, 64, 32, lane);

    // ---- O += P @ V : B[t,d] = Vt[d, t], Vt rows are t-contiguous ----
    {
      v16bf vb[4];
      #pragma unroll
      for (int ai = 0; ai < 4; ++ai)
        vb[ai] = load_frag_ab(Vh, 16 * ai, SEQ, t0, lane);
      #pragma unroll
      for (int ai = 0; ai < 4; ++ai)
        o_acc[ai] = wmma_bf16(pa0, vb[ai], o_acc[ai]);
      #pragma unroll
      for (int ai = 0; ai < 4; ++ai)
        vb[ai] = load_frag_ab(Vh, 16 * ai, SEQ, t0 + 32, lane);
      #pragma unroll
      for (int ai = 0; ai < 4; ++ai)
        o_acc[ai] = wmma_bf16(pa1, vb[ai], o_acc[ai]);
    }
  }

  // ---- normalize and store O (bf16) back in [B,S,D] layout ----
  #pragma unroll
  for (int ai = 0; ai < 4; ++ai)
    #pragma unroll
    for (int v = 0; v < 8; ++v) {
      long mg = m0 + v + mrow;
      long dg = (long)h * HDIM + 16 * ai + ncol;
      float o = o_acc[ai][v] / l_run[v];
      Ob[((long)b * SEQ + mg) * DMODEL + dg] = (bf16_t)o;
    }
}

// ---------------------------------------------------------------
// Launcher.  Workspace layout (bf16 buffers), total 56 MB.
// ---------------------------------------------------------------
extern "C" void kernel_launch(void* const* d_in, const int* in_sizes, int n_in,
                              void* d_out, int out_size, void* d_ws, size_t ws_size,
                              hipStream_t stream) {
  const float* q  = (const float*)d_in[0];
  const float* k  = (const float*)d_in[1];
  const float* v  = (const float*)d_in[2];
  const int*   mk = (const int*)  d_in[3];
  const float* Wq = (const float*)d_in[4];
  const float* Wk = (const float*)d_in[5];
  const float* Wv = (const float*)d_in[6];
  const float* Wo = (const float*)d_in[7];
  float* out = (float*)d_out;

  char* ws = (char*)d_ws;
  bf16_t* qi  = (bf16_t*)(ws);
  bf16_t* ki  = (bf16_t*)(ws + (8ull  << 20));
  bf16_t* vi  = (bf16_t*)(ws + (16ull << 20));
  bf16_t* wqb = (bf16_t*)(ws + (24ull << 20));
  bf16_t* wkb = (bf16_t*)(ws + (26ull << 20));
  bf16_t* wvb = (bf16_t*)(ws + (28ull << 20));
  bf16_t* wob = (bf16_t*)(ws + (30ull << 20));
  bf16_t* Qb  = (bf16_t*)(ws + (32ull << 20));
  bf16_t* Kb  = (bf16_t*)(ws + (40ull << 20));
  bf16_t* Vb  = (bf16_t*)(ws + (48ull << 20));
  bf16_t* Vt  = vi;   // vi dead after V projection
  bf16_t* Ob  = qi;   // qi dead after Q projection

  const int nAct = BATCH * SEQ * DMODEL;   // 4,194,304
  const int nW   = DMODEL * DMODEL;        // 1,048,576

  // 1) casts fp32 -> bf16
  cast_f32_bf16_x4<<<nAct / 4 / 256, 256, 0, stream>>>(q,  qi,  nAct / 4);
  cast_f32_bf16_x4<<<nAct / 4 / 256, 256, 0, stream>>>(k,  ki,  nAct / 4);
  cast_f32_bf16_x4<<<nAct / 4 / 256, 256, 0, stream>>>(v,  vi,  nAct / 4);
  cast_f32_bf16_x4<<<nW   / 4 / 256, 256, 0, stream>>>(Wq, wqb, nW / 4);
  cast_f32_bf16_x4<<<nW   / 4 / 256, 256, 0, stream>>>(Wk, wkb, nW / 4);
  cast_f32_bf16_x4<<<nW   / 4 / 256, 256, 0, stream>>>(Wv, wvb, nW / 4);
  cast_f32_bf16_x4<<<nW   / 4 / 256, 256, 0, stream>>>(Wo, wob, nW / 4);

  // 2) Q/K/V projections (bf16 out)
  dim3 ggrid(DMODEL / 64, MROWS / 64);  // (16, 64)
  gemm_nt_kernel<false><<<ggrid, 128, 0, stream>>>(qi, wqb, Qb, nullptr);
  gemm_nt_kernel<false><<<ggrid, 128, 0, stream>>>(ki, wkb, Kb, nullptr);
  gemm_nt_kernel<false><<<ggrid, 128, 0, stream>>>(vi, wvb, Vb, nullptr);

  // 3) V transpose to [B,H,DK,S]
  transpose_v_kernel<<<nAct / 256, 256, 0, stream>>>(Vb, Vt);

  // 4) flash attention
  dim3 fgrid(SEQ / 64, NHEAD, BATCH);   // (32, 16, 2)
  flash_attn_kernel<<<fgrid, 128, 0, stream>>>(Qb, Kb, Vt, mk, Ob);

  // 5) output projection (fp32 out)
  gemm_nt_kernel<true><<<ggrid, 128, 0, stream>>>(Ob, wob, nullptr, out);
}
